// TRA_57999238365459
// MI455X (gfx1250) — matrix operand order repaired and checked by reading, not compiled, not run on hardware
//
#include <hip/hip_runtime.h>
#include <cstdint>

// ---------------- CDNA5 WMMA types ----------------
typedef __attribute__((ext_vector_type(16))) __bf16 v16bf;
typedef __attribute__((ext_vector_type(8)))  float  v8f;

#define HW_    65536      // 256*256
#define CIN_   192
#define NPIX_  262144     // 4*256*256
#define NH_    6
#define HD_    32
#define QKVC_  576        // 3*192

// bf16 helpers (manual bit ops -> no dependence on __bf16 arithmetic)
__device__ __forceinline__ unsigned short f32_to_bf16(float f) {
  union { float f; uint32_t u; } c; c.f = f;
  uint32_t u = c.u;
  uint32_t r = u + 0x7FFFu + ((u >> 16) & 1u);   // round-to-nearest-even
  return (unsigned short)(r >> 16);
}
__device__ __forceinline__ float bf16_to_f32(unsigned short h) {
  union { uint32_t u; float f; } c; c.u = ((uint32_t)h) << 16;
  return c.f;
}

// 16 bf16 elements = one WMMA A/B fragment per lane (two 16B chunks)
union Frag {
  v16bf bf;
  unsigned short us[16];
  uint4 q[2];
};

// CDNA5 async global->LDS copy of one 16B chunk per lane (ASYNCcnt-tracked)
__device__ __forceinline__ void async_copy_b128(uint32_t lds_addr, uint32_t byte_off,
                                                const void* sbase) {
  asm volatile("global_load_async_to_lds_b128 %0, %1, %2"
               :: "v"(lds_addr), "v"(byte_off), "s"(sbase) : "memory");
}
__device__ __forceinline__ void async_wait0() {
  asm volatile("s_wait_asynccnt 0" ::: "memory");
}

// ---------------- kernel 0: convert weights fp32 -> bf16 ----------------
__global__ __launch_bounds__(256) void k_convert_w(
    const float* __restrict__ qkv_w, const float* __restrict__ proj_w,
    unsigned short* __restrict__ wq, unsigned short* __restrict__ wp) {
  int i = blockIdx.x * 256 + threadIdx.x;
  if (i < QKVC_ * CIN_) wq[i] = f32_to_bf16(qkv_w[i]);
  if (i < CIN_ * CIN_)  wp[i] = f32_to_bf16(proj_w[i]);
}

// ------ kernel 1: x0 (B,C,H,W) fp32 -> x_t [pixel][192] bf16 (pixel-major) ------
__global__ __launch_bounds__(256) void k_transpose_x(
    const float* __restrict__ x0, unsigned short* __restrict__ x_t) {
  int p = blockIdx.x * 256 + threadIdx.x;          // 0..NPIX-1
  int b = p >> 16;
  int r = p & 65535;
  const float* src = x0 + (size_t)b * CIN_ * HW_ + r;
  unsigned short* dst = x_t + (size_t)p * CIN_;
#pragma unroll
  for (int cc = 0; cc < CIN_ / 16; ++cc) {
    unsigned short t[16];
#pragma unroll
    for (int i = 0; i < 16; ++i)
      t[i] = f32_to_bf16(src[(size_t)(cc * 16 + i) * HW_]);   // coalesced across lanes
    uint32_t w[8];
#pragma unroll
    for (int i = 0; i < 8; ++i)
      w[i] = (uint32_t)t[2 * i] | ((uint32_t)t[2 * i + 1] << 16);
    uint4* d = (uint4*)(dst + cc * 16);
    d[0] = make_uint4(w[0], w[1], w[2], w[3]);
    d[1] = make_uint4(w[4], w[5], w[6], w[7]);
  }
}

// -------- kernel 2: QKV GEMM: qkv_t[pixel][576] = qkv_w(576x192) * x + qkv_b --------
// block = 256 thr (8 waves); each wave owns 64 pixels, async-stages its 24KB
// activation slab into LDS ONCE, then loops all 36 M-tiles (x_t read 1x from HBM).
__global__ __launch_bounds__(256) void k_qkv_gemm(
    const unsigned short* __restrict__ wq, const unsigned short* __restrict__ x_t,
    const float* __restrict__ qkv_b, unsigned short* __restrict__ qkv_t) {
  extern __shared__ unsigned short smem[];          // 8 waves * 64*192 u16 = 192KB
  const int lane = threadIdx.x & 31;
  const int wave = threadIdx.x >> 5;
  const int mrow = lane & 15;
  const int half = lane >> 4;
  const int Nwave = blockIdx.x * 512 + wave * 64;   // 64 pixels per wave

  unsigned short* lb = smem + (size_t)wave * 64 * CIN_;

  // ---- async stage: 64 rows x 192 ch = one contiguous 24KB global span -> LDS ----
  {
    const uint32_t goff0 = (uint32_t)Nwave * (CIN_ * 2);   // byte offset into x_t
    const uint32_t lds0 = (uint32_t)(uintptr_t)lb;         // LDS byte address
#pragma unroll
    for (int i = 0; i < 48; ++i) {                         // 48*32 lanes*16B = 24KB
      uint32_t c = (uint32_t)(lane + 32 * i) * 16u;
      async_copy_b128(lds0 + c, goff0 + c, (const void*)x_t);
    }
    async_wait0();   // wave reads only its own region: no barrier needed
  }

  // ---- loop all M tiles; B fragments come from LDS, A from global (L2-resident) ----
  for (int mt = 0; mt < QKVC_ / 16; ++mt) {
    const int Mbase = mt * 16;
    v8f acc[4] = {};
    const unsigned short* arow = wq + (size_t)(Mbase + mrow) * CIN_;
#pragma unroll
    for (int k0 = 0; k0 < CIN_; k0 += 32) {
      Frag a;
      a.q[0] = *(const uint4*)(arow + k0 + half * 8);
      a.q[1] = *(const uint4*)(arow + k0 + 16 + half * 8);
#pragma unroll
      for (int j = 0; j < 4; ++j) {
        const unsigned short* brow = lb + (j * 16 + mrow) * CIN_ + k0 + half * 8;
        Frag b;
        b.q[0] = *(const uint4*)(brow);
        b.q[1] = *(const uint4*)(brow + 16);
        acc[j] = __builtin_amdgcn_wmma_f32_16x16x32_bf16(
            false, a.bf, false, b.bf, (short)0, acc[j], false, false);
      }
    }
    // bias + bf16 store: lane's 8 values = 8 consecutive channels at one pixel
    float bv[8];
#pragma unroll
    for (int r = 0; r < 8; ++r) bv[r] = qkv_b[Mbase + half * 8 + r];
#pragma unroll
    for (int j = 0; j < 4; ++j) {
      int n = Nwave + j * 16 + mrow;
      unsigned short* dst = qkv_t + (size_t)n * QKVC_ + Mbase + half * 8;
      unsigned short t[8];
#pragma unroll
      for (int r = 0; r < 8; ++r) t[r] = f32_to_bf16(acc[j][r] + bv[r]);
      uint32_t w[4];
#pragma unroll
      for (int i = 0; i < 4; ++i)
        w[i] = (uint32_t)t[2 * i] | ((uint32_t)t[2 * i + 1] << 16);
      *(uint4*)dst = make_uint4(w[0], w[1], w[2], w[3]);
    }
  }
}

// -------- kernel 3: window attention, one wave per (window, head) --------
__global__ __launch_bounds__(32) void k_attn(
    const unsigned short* __restrict__ qkv_t, const float* __restrict__ rel_table,
    const float* __restrict__ logit_scale, unsigned short* __restrict__ o_t) {
  __shared__ __align__(16) unsigned short attn_s[16][16];  // [query][key] bf16
  __shared__ __align__(16) unsigned short v_s[32][16];     // [channel][key] bf16

  const int lane = threadIdx.x;
  const int wid = blockIdx.x;             // 0 .. 16384*6-1
  const int head = wid % NH_;
  const int win = wid / NH_;
  const int wn = win & 63;
  const int hn = (win >> 6) & 63;
  const int b  = win >> 12;
  const int n16 = lane & 15;
  const int half = lane >> 4;

  const size_t winbase = ((size_t)b * HW_ + (size_t)(hn * 4) * 256 + wn * 4);
  auto rowbase = [&](int m) -> size_t {
    return (winbase + (size_t)(m >> 2) * 256 + (m & 3)) * QKVC_ + head * 96;
  };

  // ---- q fragment (A layout) with cosine normalization ----
  size_t qb = rowbase(n16);
  Frag qa;
  qa.q[0] = *(const uint4*)(qkv_t + qb + half * 8);
  qa.q[1] = *(const uint4*)(qkv_t + qb + 16 + half * 8);
  {
    float v[16], ss = 0.f;
#pragma unroll
    for (int i = 0; i < 16; ++i) { v[i] = bf16_to_f32(qa.us[i]); ss += v[i] * v[i]; }
    ss += __shfl_xor(ss, 16, 32);                    // other half of the 32-dim row
    float inv = 1.0f / fmaxf(sqrtf(ss), 1e-12f);
#pragma unroll
    for (int i = 0; i < 16; ++i) qa.us[i] = f32_to_bf16(v[i] * inv);
  }

  // ---- k fragment (B layout: lane n holds key-row n, same K order as q) ----
  size_t kb = rowbase(n16) + 32 + half * 8;
  Frag kf;
  kf.q[0] = *(const uint4*)(qkv_t + kb);
  kf.q[1] = *(const uint4*)(qkv_t + kb + 16);
  {
    float v[16], ss = 0.f;
#pragma unroll
    for (int i = 0; i < 16; ++i) { v[i] = bf16_to_f32(kf.us[i]); ss += v[i] * v[i]; }
    ss += __shfl_xor(ss, 16, 32);
    float inv = 1.0f / fmaxf(sqrtf(ss), 1e-12f);
#pragma unroll
    for (int i = 0; i < 16; ++i) kf.us[i] = f32_to_bf16(v[i] * inv);
  }

  // ---- S = qn * kn^T  (16x16, K=32) ----
  v8f S = {};
  S = __builtin_amdgcn_wmma_f32_16x16x32_bf16(false, qa.bf, false, kf.bf,
                                              (short)0, S, false, false);

  // ---- scale + relative-position bias + softmax over keys ----
  const float scale = __expf(fminf(logit_scale[head], 4.60517019f)); // log(100)
  float pr[8];
#pragma unroll
  for (int r = 0; r < 8; ++r) {
    int m = r + 8 * half;                            // query index
    int ridx = ((m >> 2) - (n16 >> 2) + 3) * 7 + ((m & 3) - (n16 & 3) + 3);
    float lg = S[r] * scale + rel_table[ridx * NH_ + head];
    float mx = lg;
#pragma unroll
    for (int msk = 1; msk < 16; msk <<= 1) mx = fmaxf(mx, __shfl_xor(mx, msk, 32));
    float e = __expf(lg - mx);
    float sm = e;
#pragma unroll
    for (int msk = 1; msk < 16; msk <<= 1) sm += __shfl_xor(sm, msk, 32);
    pr[r] = e / sm;
  }

  // ---- restage attn (bf16) and v (transposed) through LDS ----
#pragma unroll
  for (int r = 0; r < 8; ++r) attn_s[r + 8 * half][n16] = f32_to_bf16(pr[r]);
  {
    size_t vb = rowbase(n16) + 64 + half * 16;       // v row n16, 16 channels
    Frag vv;
    vv.q[0] = *(const uint4*)(qkv_t + vb);
    vv.q[1] = *(const uint4*)(qkv_t + vb + 8);
#pragma unroll
    for (int i = 0; i < 16; ++i) v_s[half * 16 + i][n16] = vv.us[i];
  }
  __syncthreads();   // single-wave WG: lowers to S_NOP, orders LDS

  // ---- O = attn(16x16, K padded to 32) * v(16x32) : two N-halves ----
  Frag a2;
  a2.q[0] = *(const uint4*)(&attn_s[n16][half * 8]);  // K = half*8..+7
  a2.q[1] = make_uint4(0, 0, 0, 0);                   // K = 16..31 pad
#pragma unroll
  for (int j2 = 0; j2 < 2; ++j2) {
    Frag b2;
    const unsigned short* vrow = &v_s[j2 * 16 + n16][0];
    b2.q[0] = *(const uint4*)(vrow + half * 8);
    b2.q[1] = make_uint4(0, 0, 0, 0);
    v8f o = {};
    o = __builtin_amdgcn_wmma_f32_16x16x32_bf16(false, a2.bf, false, b2.bf,
                                                (short)0, o, false, false);
    int ch = head * HD_ + j2 * 16 + n16;
#pragma unroll
    for (int r = 0; r < 8; ++r) {
      int m = r + 8 * half;
      size_t po = winbase + (size_t)(m >> 2) * 256 + (m & 3);
      o_t[po * CIN_ + ch] = f32_to_bf16(o[r]);
    }
  }
}

// -------- kernel 4: proj GEMM: out(B,C,H,W) fp32 = proj_w(192x192) * o + proj_b --------
// Same async-LDS staged structure: o_t read exactly once from HBM.
__global__ __launch_bounds__(256) void k_proj_gemm(
    const unsigned short* __restrict__ wp, const unsigned short* __restrict__ o_t,
    const float* __restrict__ proj_b, float* __restrict__ out) {
  extern __shared__ unsigned short smem[];
  const int lane = threadIdx.x & 31;
  const int wave = threadIdx.x >> 5;
  const int mrow = lane & 15;
  const int half = lane >> 4;
  const int Nwave = blockIdx.x * 512 + wave * 64;

  unsigned short* lb = smem + (size_t)wave * 64 * CIN_;

  {
    const uint32_t goff0 = (uint32_t)Nwave * (CIN_ * 2);
    const uint32_t lds0 = (uint32_t)(uintptr_t)lb;
#pragma unroll
    for (int i = 0; i < 48; ++i) {
      uint32_t c = (uint32_t)(lane + 32 * i) * 16u;
      async_copy_b128(lds0 + c, goff0 + c, (const void*)o_t);
    }
    async_wait0();
  }

  for (int mt = 0; mt < CIN_ / 16; ++mt) {   // 12 M tiles
    const int Mbase = mt * 16;
    v8f acc[4] = {};
    const unsigned short* arow = wp + (size_t)(Mbase + mrow) * CIN_;
#pragma unroll
    for (int k0 = 0; k0 < CIN_; k0 += 32) {
      Frag a;
      a.q[0] = *(const uint4*)(arow + k0 + half * 8);
      a.q[1] = *(const uint4*)(arow + k0 + 16 + half * 8);
#pragma unroll
      for (int j = 0; j < 4; ++j) {
        const unsigned short* brow = lb + (j * 16 + mrow) * CIN_ + k0 + half * 8;
        Frag b;
        b.q[0] = *(const uint4*)(brow);
        b.q[1] = *(const uint4*)(brow + 16);
        acc[j] = __builtin_amdgcn_wmma_f32_16x16x32_bf16(
            false, a.bf, false, b.bf, (short)0, acc[j], false, false);
      }
    }
    float bv[8];
#pragma unroll
    for (int r = 0; r < 8; ++r) bv[r] = proj_b[Mbase + half * 8 + r];
#pragma unroll
    for (int j = 0; j < 4; ++j) {
      int n = Nwave + j * 16 + mrow;
      int bb = n >> 16, rr = n & 65535;
#pragma unroll
      for (int r = 0; r < 8; ++r) {
        int c = Mbase + half * 8 + r;
        out[((size_t)(bb * CIN_ + c) << 16) + rr] = acc[j][r] + bv[r];
      }
    }
  }
}

// ---------------- host side ----------------
extern "C" void kernel_launch(void* const* d_in, const int* in_sizes, int n_in,
                              void* d_out, int out_size, void* d_ws, size_t ws_size,
                              hipStream_t stream) {
  const float* x0          = (const float*)d_in[0];
  const float* qkv_w       = (const float*)d_in[1];
  const float* qkv_b       = (const float*)d_in[2];
  const float* proj_w      = (const float*)d_in[3];
  const float* proj_b      = (const float*)d_in[4];
  const float* rel_table   = (const float*)d_in[5];
  const float* logit_scale = (const float*)d_in[6];
  float* out = (float*)d_out;

  // workspace layout (all bf16 as u16 bits)
  unsigned short* wq    = (unsigned short*)d_ws;          // 576*192
  unsigned short* wp    = wq + QKVC_ * CIN_;              // 192*192
  unsigned short* x_t   = wp + CIN_ * CIN_;               // NPIX*192  (pixel-major)
  unsigned short* o_t   = x_t;                            // alias: x_t dead after QKV GEMM
  unsigned short* qkv_t = x_t + (size_t)NPIX_ * CIN_;     // NPIX*576  (pixel-major)

  const size_t lds_bytes = 8u * 64u * CIN_ * 2u;          // 192KB (CDNA5 WGP: 320KB)

  k_convert_w<<<(QKVC_ * CIN_ + 255) / 256, 256, 0, stream>>>(qkv_w, proj_w, wq, wp);
  k_transpose_x<<<NPIX_ / 256, 256, 0, stream>>>(x0, x_t);
  k_qkv_gemm<<<NPIX_ / 512, 256, lds_bytes, stream>>>(wq, x_t, qkv_b, qkv_t);
  k_attn<<<(NPIX_ / 16) * NH_, 32, 0, stream>>>(qkv_t, rel_table, logit_scale, o_t);
  k_proj_gemm<<<NPIX_ / 512, 256, lds_bytes, stream>>>(wp, o_t, proj_b, out);
}
